// QuantumDecoder_1726576853348
// MI455X (gfx1250) — matrix operand order repaired and checked by reading, not compiled, not run on hardware
//
#include <hip/hip_runtime.h>

// ---------------------------------------------------------------------------
// QuantumDecoder: out = Re(conj(s) * (s @ M)), s complex [16384,1024], M [1024,1024]
// Complex GEMM via bf16x3-split WMMA (v_wmma_f32_16x16x32_bf16).
// Both operands are pre-split/packed into WMMA fragment layout by pack kernels
// so the GEMM inner loop is loads + WMMA only (no per-chunk VALU conversion).
// ---------------------------------------------------------------------------

typedef __attribute__((ext_vector_type(16))) __bf16 v16bf;
typedef __attribute__((ext_vector_type(8)))  float  v8f;

#define D_DIM 1024          // state dim (= N = K of the GEMM)
#define MT_N  1024          // M tiles of 16 rows (16384/16)
#define KT_N  32            // K chunks of 32
#define NT_N  64            // N tiles of 16
#define FRAG_DW 8           // dwords per lane per fragment

union Frag {
    v16bf          bf;
    unsigned       u32[8];
    unsigned short us[16];
};

__device__ __forceinline__ unsigned short bf16_rne(float x) {
    unsigned u = __float_as_uint(x);
    u += 0x7FFFu + ((u >> 16) & 1u);          // round-to-nearest-even
    return (unsigned short)(u >> 16);
}

// fp32 -> (bf16 hi, bf16 lo) split: x ~= hi + lo, |lo| <~ 2^-8 |x|
__device__ __forceinline__ void bsplit(float x, unsigned short& h, unsigned short& l) {
    unsigned short hs = bf16_rne(x);
    float hf = __uint_as_float((unsigned)hs << 16);
    h = hs;
    l = bf16_rne(x - hf);
}

__device__ __forceinline__ void split4(const float4 f, Frag& h, Frag& l, int base) {
    bsplit(f.x, h.us[base + 0], l.us[base + 0]);
    bsplit(f.y, h.us[base + 1], l.us[base + 1]);
    bsplit(f.z, h.us[base + 2], l.us[base + 2]);
    bsplit(f.w, h.us[base + 3], l.us[base + 3]);
}

__device__ __forceinline__ void loadFrag(Frag& f, const unsigned* __restrict__ p) {
    *(uint4*)&f.u32[0] = *(const uint4*)(p);
    *(uint4*)&f.u32[4] = *(const uint4*)(p + 4);
}

__device__ __forceinline__ void storeFrag(const Frag& f, unsigned* __restrict__ p) {
    *(uint4*)(p)     = *(const uint4*)&f.u32[0];
    *(uint4*)(p + 4) = *(const uint4*)&f.u32[4];
}

__device__ __forceinline__ v8f wmma_bf16(const Frag& a, const Frag& b, v8f c) {
    // 8 args: (neg_a, A, neg_b, B, c_mod, C, reuse_a, reuse_b)
    return __builtin_amdgcn_wmma_f32_16x16x32_bf16(false, a.bf, false, b.bf,
                                                   (short)0, c, false, false);
}

// ---------------------------------------------------------------------------
// Pack M_re / M_im into B-operand fragment layout (bf16 32x16 per fragment):
//   lanes 0-15 : K = 0..7 & 16..23 of chunk (2 bf16 per dword, K-major)
//   lanes 16-31: K = 8..15 & 24..31 of chunk
// Buffers: reH reL imH imL imHn imLn ; each [kt=32][nt=64][lane=32][dw=8]
// imHn/imLn are sign-flipped so accR accumulates (-sim) x Mim with no VALU.
// ---------------------------------------------------------------------------
__global__ void __launch_bounds__(256)
qd_pack_M(const float* __restrict__ Mre, const float* __restrict__ Mim,
          unsigned* __restrict__ reH, unsigned* __restrict__ reL,
          unsigned* __restrict__ imH, unsigned* __restrict__ imL,
          unsigned* __restrict__ imHn, unsigned* __restrict__ imLn) {
    const int lane = threadIdx.x & 31;
    const int wave = threadIdx.x >> 5;
    const int frag = blockIdx.x * 8 + wave;        // 0..2047 == kt*64 + nt
    const int kt = frag >> 6;
    const int nt = frag & 63;
    const int n  = nt * 16 + (lane & 15);          // column (N)
    const int kb = kt * 32 + (lane >> 4) * 8;      // first K this lane holds
    const size_t off = ((size_t)frag * 32 + lane) * FRAG_DW;

#pragma unroll
    for (int v = 0; v < 8; ++v) {
        const size_t i0 = (size_t)(kb + 2 * v + (v >= 4 ? 8 : 0)) * D_DIM + n;   // row-major M[K][N]
        const size_t i1 = i0 + D_DIM;
        unsigned short h0, l0, h1, l1;
        bsplit(Mre[i0], h0, l0); bsplit(Mre[i1], h1, l1);
        reH[off + v] = (unsigned)h0 | ((unsigned)h1 << 16);
        reL[off + v] = (unsigned)l0 | ((unsigned)l1 << 16);
        bsplit(Mim[i0], h0, l0); bsplit(Mim[i1], h1, l1);
        const unsigned ih = (unsigned)h0 | ((unsigned)h1 << 16);
        const unsigned il = (unsigned)l0 | ((unsigned)l1 << 16);
        imH[off + v]  = ih;
        imL[off + v]  = il;
        imHn[off + v] = ih ^ 0x80008000u;          // negate both packed bf16
        imLn[off + v] = il ^ 0x80008000u;
    }
}

// ---------------------------------------------------------------------------
// Pack state_re / state_im into A-operand fragment layout (bf16 16x32):
//   lane = row within 16-row tile; lanes 0-15 hold K 0..7 & 16..23,
//   lanes 16-31 hold K 8..15 & 24..31 (2 bf16 per dword).
// Buffers: aReH aReL aImH aImL ; each [mt=1024][kt=32][lane=32][dw=8]
// One wave builds one (mt, kt) fragment for all four buffers.
// ---------------------------------------------------------------------------
__global__ void __launch_bounds__(256)
qd_pack_S(const float* __restrict__ sre, const float* __restrict__ sim,
          unsigned* __restrict__ aReH, unsigned* __restrict__ aReL,
          unsigned* __restrict__ aImH, unsigned* __restrict__ aImL) {
    const int lane = threadIdx.x & 31;
    const int wave = threadIdx.x >> 5;
    const int frag = blockIdx.x * 8 + wave;        // 0..32767 == mt*32 + kt
    const int mt = frag >> 5;
    const int kt = frag & 31;
    const int row  = mt * 16 + (lane & 15);
    const int koff = kt * 32 + (lane >> 4) * 8;    // 0 or 8 within chunk
    const float* pr = sre + (size_t)row * D_DIM + koff;
    const float* pi = sim + (size_t)row * D_DIM + koff;

    Frag rH, rL, iH, iL;
    // us[0..7] = K koff..koff+7 ; us[8..15] = K koff+16..koff+23
    split4(*(const float4*)(pr),      rH, rL, 0);
    split4(*(const float4*)(pr + 4),  rH, rL, 4);
    split4(*(const float4*)(pr + 16), rH, rL, 8);
    split4(*(const float4*)(pr + 20), rH, rL, 12);
    split4(*(const float4*)(pi),      iH, iL, 0);
    split4(*(const float4*)(pi + 4),  iH, iL, 4);
    split4(*(const float4*)(pi + 16), iH, iL, 8);
    split4(*(const float4*)(pi + 20), iH, iL, 12);

    const size_t off = ((size_t)frag * 32 + lane) * FRAG_DW;
    storeFrag(rH, aReH + off);
    storeFrag(rL, aReL + off);
    storeFrag(iH, aImH + off);
    storeFrag(iL, aImL + off);
}

// ---------------------------------------------------------------------------
// Main GEMM: workgroup tile 64(M) x 128(N), 8 waves, wave tile 32x32.
// yre = sre@Mre - sim@Mim ; yim = sre@Mim + sim@Mre ; out = sre*yre + sim*yim
// Each real product uses bf16x3: hi*hi + hi*lo + lo*hi, fp32 accumulate.
// Inner loop: 16 A-frag b128 loads + 24 B-frag b128 loads + 48 WMMAs.
// ---------------------------------------------------------------------------
__global__ void __launch_bounds__(256)
qd_gemm(const float* __restrict__ sre, const float* __restrict__ sim,
        const unsigned* __restrict__ reH, const unsigned* __restrict__ reL,
        const unsigned* __restrict__ imH, const unsigned* __restrict__ imL,
        const unsigned* __restrict__ imHn, const unsigned* __restrict__ imLn,
        const unsigned* __restrict__ aReH, const unsigned* __restrict__ aReL,
        const unsigned* __restrict__ aImH, const unsigned* __restrict__ aImL,
        float* __restrict__ out) {
    const int lane  = threadIdx.x & 31;
    const int wave  = threadIdx.x >> 5;
    const int waveM = blockIdx.y * 64  + (wave & 1) * 32;   // output rows
    const int waveN = blockIdx.x * 128 + (wave >> 1) * 32;  // output cols

    v8f accR[2][2], accI[2][2];
#pragma unroll
    for (int i = 0; i < 2; ++i)
#pragma unroll
        for (int j = 0; j < 2; ++j)
#pragma unroll
            for (int e = 0; e < 8; ++e) { accR[i][j][e] = 0.f; accI[i][j][e] = 0.f; }

    // Per-lane base offsets (advance by FRAG_STEP per kt)
    const size_t FRAG_STEP = 32u * FRAG_DW;                       // dwords per kt
    const size_t aBase0 = (((size_t)((waveM >> 4) + 0) * KT_N) * 32 + lane) * FRAG_DW;
    const size_t aBase1 = (((size_t)((waveM >> 4) + 1) * KT_N) * 32 + lane) * FRAG_DW;
    const size_t bBase0 = (((size_t)((waveN >> 4) + 0)) * 32 + lane) * FRAG_DW;
    const size_t bBase1 = (((size_t)((waveN >> 4) + 1)) * 32 + lane) * FRAG_DW;
    const size_t B_STEP = (size_t)NT_N * 32 * FRAG_DW;            // dwords per kt

    for (int kt = 0; kt < KT_N; ++kt) {
        const size_t a0 = aBase0 + (size_t)kt * FRAG_STEP;
        const size_t a1 = aBase1 + (size_t)kt * FRAG_STEP;
        const size_t b0 = bBase0 + (size_t)kt * B_STEP;
        const size_t b1 = bBase1 + (size_t)kt * B_STEP;

        Frag srH[2], srL[2], siH[2], siL[2];
        loadFrag(srH[0], aReH + a0);  loadFrag(srL[0], aReL + a0);
        loadFrag(siH[0], aImH + a0);  loadFrag(siL[0], aImL + a0);
        loadFrag(srH[1], aReH + a1);  loadFrag(srL[1], aReL + a1);
        loadFrag(siH[1], aImH + a1);  loadFrag(siL[1], aImL + a1);

#pragma unroll
        for (int ni = 0; ni < 2; ++ni) {
            const size_t boff = (ni == 0) ? b0 : b1;
            Frag brH, brL, biH, biL, biHn, biLn;
            loadFrag(brH,  reH  + boff);
            loadFrag(brL,  reL  + boff);
            loadFrag(biH,  imH  + boff);
            loadFrag(biL,  imL  + boff);
            loadFrag(biHn, imHn + boff);
            loadFrag(biLn, imLn + boff);

#pragma unroll
            for (int mi = 0; mi < 2; ++mi) {
                v8f r = accR[mi][ni];
                v8f c = accI[mi][ni];
                // Re(y): sre@Mre  +  (-sim)@Mim   (bf16x3 each)
                r = wmma_bf16(srH[mi], brH,  r);
                r = wmma_bf16(srH[mi], brL,  r);
                r = wmma_bf16(srL[mi], brH,  r);
                r = wmma_bf16(siH[mi], biHn, r);
                r = wmma_bf16(siH[mi], biLn, r);
                r = wmma_bf16(siL[mi], biHn, r);
                // Im(y): sre@Mim  +  sim@Mre
                c = wmma_bf16(srH[mi], biH,  c);
                c = wmma_bf16(srH[mi], biL,  c);
                c = wmma_bf16(srL[mi], biH,  c);
                c = wmma_bf16(siH[mi], brH,  c);
                c = wmma_bf16(siH[mi], brL,  c);
                c = wmma_bf16(siL[mi], brH,  c);
                accR[mi][ni] = r;
                accI[mi][ni] = c;
            }
        }
    }

    // ---- fused epilogue: out = sre*Re(y) + sim*Im(y)
    // C/D layout: VGPR j -> row j (lanes 0-15) / row j+8 (lanes 16-31), col = lane&15
#pragma unroll
    for (int mi = 0; mi < 2; ++mi) {
#pragma unroll
        for (int ni = 0; ni < 2; ++ni) {
            const int r0  = waveM + mi * 16 + (lane >> 4) * 8;
            const int col = waveN + ni * 16 + (lane & 15);
#pragma unroll
            for (int j = 0; j < 8; ++j) {
                const size_t idx = (size_t)(r0 + j) * D_DIM + col;
                out[idx] = sre[idx] * accR[mi][ni][j] + sim[idx] * accI[mi][ni][j];
            }
        }
    }
}

// ---------------------------------------------------------------------------
extern "C" void kernel_launch(void* const* d_in, const int* in_sizes, int n_in,
                              void* d_out, int out_size, void* d_ws, size_t ws_size,
                              hipStream_t stream) {
    const float* sre = (const float*)d_in[0];   // [8,2048,1024] f32
    const float* sim = (const float*)d_in[1];
    const float* Mre = (const float*)d_in[2];   // [1024,1024] f32
    const float* Mim = (const float*)d_in[3];
    float* out = (float*)d_out;

    // Workspace: 6 B-buffers (2 MB each) + 4 A-buffers (33.5 MB each) ~ 147 MB
    const size_t bufDwB = (size_t)KT_N * NT_N * 32 * FRAG_DW;   // 524288 dwords
    const size_t bufDwA = (size_t)MT_N * KT_N * 32 * FRAG_DW;   // 8388608 dwords
    unsigned* ws   = (unsigned*)d_ws;
    unsigned* reH  = ws + 0 * bufDwB;
    unsigned* reL  = ws + 1 * bufDwB;
    unsigned* imH  = ws + 2 * bufDwB;
    unsigned* imL  = ws + 3 * bufDwB;
    unsigned* imHn = ws + 4 * bufDwB;
    unsigned* imLn = ws + 5 * bufDwB;
    unsigned* aReH = ws + 6 * bufDwB;
    unsigned* aReL = aReH + bufDwA;
    unsigned* aImH = aReL + bufDwA;
    unsigned* aImL = aImH + bufDwA;

    // B pack: 2048 fragments, one wave each -> 256 blocks of 8 waves
    qd_pack_M<<<256, 256, 0, stream>>>(Mre, Mim, reH, reL, imH, imL, imHn, imLn);

    // A pack: 32768 fragments, one wave each -> 4096 blocks of 8 waves
    qd_pack_S<<<4096, 256, 0, stream>>>(sre, sim, aReH, aReL, aImH, aImL);

    // GEMM grid: x = N/128 = 8, y = (B*S)/64 = 256
    qd_gemm<<<dim3(8, 256), 256, 0, stream>>>(sre, sim,
                                              reH, reL, imH, imL, imHn, imLn,
                                              aReH, aReL, aImH, aImL, out);
}
